// SelfAttention_1675037245464
// MI455X (gfx1250) — matrix operand order repaired and checked by reading, not compiled, not run on hardware
//
#include <hip/hip_runtime.h>

typedef unsigned int u32;
typedef unsigned short u16;
typedef __attribute__((ext_vector_type(16))) __bf16 bf16x16;
typedef __attribute__((ext_vector_type(4)))  __bf16 bf16x4;
typedef __attribute__((ext_vector_type(8)))  float  f32x8;
typedef __attribute__((ext_vector_type(4)))  float  f32x4;
typedef __attribute__((ext_vector_type(4)))  u32    u32x4;

union Frag {
  bf16x16 v;
  u32x4   q[2];
  u16     h[16];
  bf16x4  b4[4];
};

__device__ __forceinline__ u16 f2b(float x) {
  union { __bf16 b; u16 u; } cv;
  cv.b = (__bf16)x;               // hardware f32->bf16 (RTE)
  return cv.u;
}
__device__ __forceinline__ float b2f(u16 h) {
  u32 b = ((u32)h) << 16;
  return __builtin_bit_cast(float, b);
}

// ---- WMMA fragment loaders (CDNA5 v_wmma_f32_16x16x32_bf16 layouts) ----
// A (16x32 bf16): lane L -> row m=L&15; K = h*8+{0..7} and h*8+16+{0..7}, h=L>>4
__device__ __forceinline__ bf16x16 load_a_bf16(const u16* base, int ld, int lane) {
  int r = lane & 15, hh = lane >> 4;
  const u16* p = base + (size_t)r * ld + hh * 8;
  Frag f;
  f.q[0] = *(const u32x4*)(p);
  f.q[1] = *(const u32x4*)(p + 16);
  return f.v;
}
// A fragment built from a row-major f32 source, converting to bf16
__device__ __forceinline__ bf16x16 load_a_f32(const float* base, int ld, int lane) {
  int r = lane & 15, hh = lane >> 4;
  const float* p = base + (size_t)r * ld + hh * 8;
  Frag f;
  f.b4[0] = __builtin_convertvector(*(const f32x4*)(p),      bf16x4);
  f.b4[1] = __builtin_convertvector(*(const f32x4*)(p + 4),  bf16x4);
  f.b4[2] = __builtin_convertvector(*(const f32x4*)(p + 16), bf16x4);
  f.b4[3] = __builtin_convertvector(*(const f32x4*)(p + 20), bf16x4);
  return f.v;
}
// B (32x16 bf16): lane L -> col n=L&15; K = h*8+{0..7} and h*8+16+{0..7}, h=L>>4
__device__ __forceinline__ bf16x16 load_b_bf16(const u16* base, int ld, int lane) {
  int c = lane & 15, hh = lane >> 4;
  const u16* p = base + (size_t)(hh * 8) * ld + c;
  Frag f;
#pragma unroll
  for (int s = 0; s < 8; ++s) {
    f.h[s]     = p[(size_t)s * ld];
    f.h[s + 8] = p[(size_t)(s + 16) * ld];
  }
  return f.v;
}
__device__ __forceinline__ f32x8 wmma_bf16(bf16x16 a, bf16x16 b, f32x8 c) {
  return __builtin_amdgcn_wmma_f32_16x16x32_bf16(false, a, false, b, (short)0, c,
                                                 false, false);
}

// -------- Kernel 1: spectral norm + pack weight to bf16 (into ldOut/colOff) --
__global__ void spectral_pack(const float* __restrict__ W, const float* __restrict__ u,
                              u16* __restrict__ Wn, int K, int C, int ldOut, int colOff) {
  __shared__ float sv[512];
  __shared__ float st[512];
  __shared__ float red[256];
  __shared__ float ssig;
  int tid = threadIdx.x;
  // v = l2normalize(u @ W^T)
  for (int i = tid; i < K; i += 256) {
    float s = 0.f;
    for (int j = 0; j < C; ++j) s += u[j] * W[(size_t)i * C + j];
    sv[i] = s;
  }
  __syncthreads();
  float loc = 0.f;
  for (int i = tid; i < K; i += 256) loc += sv[i] * sv[i];
  red[tid] = loc; __syncthreads();
  for (int s = 128; s > 0; s >>= 1) { if (tid < s) red[tid] += red[tid + s]; __syncthreads(); }
  float vn = sqrtf(red[0]) + 1e-12f;
  __syncthreads();
  for (int i = tid; i < K; i += 256) sv[i] /= vn;
  __syncthreads();
  // t = v @ W ; sigma = t . l2normalize(t) = |t|^2 / (|t| + eps)
  for (int j = tid; j < C; j += 256) {
    float s = 0.f;
    for (int i = 0; i < K; ++i) s += sv[i] * W[(size_t)i * C + j];
    st[j] = s;
  }
  __syncthreads();
  loc = 0.f;
  for (int j = tid; j < C; j += 256) loc += st[j] * st[j];
  red[tid] = loc; __syncthreads();
  for (int s = 128; s > 0; s >>= 1) { if (tid < s) red[tid] += red[tid + s]; __syncthreads(); }
  if (tid == 0) { float tn = sqrtf(red[0]); ssig = red[0] / (tn + 1e-12f); }
  __syncthreads();
  float inv = 1.0f / ssig;
  int total = K * C;
  for (int i2 = tid; i2 < total; i2 += 256) {
    int i = i2 / C, j = i2 - i * C;           // C is a power of two
    Wn[(size_t)i * ldOut + colOff + j] = f2b(W[i2] * inv);
  }
}

// ---------- Kernel 2: fused 3-conv GEMM  Y[65536,384] = x @ Wcat[512,384] ----
// One wave = one 16-row strip x 12 N-tiles (half of the 24); acc[12] = 96 VGPRs.
__global__ void conv_gemm(const float* __restrict__ x, const u16* __restrict__ Wcat,
                          u16* __restrict__ Y) {
  int gw   = (int)((blockIdx.x * blockDim.x + threadIdx.x) >> 5);
  int lane = threadIdx.x & 31;
  int mt   = gw >> 1;                 // 0..4095 row strip
  int half = gw & 1;                  // which 12 of the 24 N-tiles
  const float* Ab = x + (size_t)mt * 16 * 512;
  const u16* Bb = Wcat + half * 192;  // column base within concatenated weights

  f32x8 z = {0.f,0.f,0.f,0.f,0.f,0.f,0.f,0.f};
  f32x8 acc[12];
#pragma unroll
  for (int t = 0; t < 12; ++t) acc[t] = z;

  for (int kt = 0; kt < 16; ++kt) {                 // K = 512
    bf16x16 a = load_a_f32(Ab + kt * 32, 512, lane);
#pragma unroll
    for (int t = 0; t < 12; ++t) {
      bf16x16 b = load_b_bf16(Bb + (size_t)(kt * 32) * 384 + t * 16, 384, lane);
      acc[t] = wmma_bf16(a, b, acc[t]);
    }
  }
  int n = lane & 15, hh = lane >> 4;
  size_t row0 = (size_t)mt * 16;
  int cb = half * 192;
#pragma unroll
  for (int t = 0; t < 12; ++t) {
#pragma unroll
    for (int v = 0; v < 8; ++v) {
      Y[(row0 + v + 8 * hh) * 384 + cb + t * 16 + n] = f2b(acc[t][v]);
    }
  }
}

// --------- Kernel 3: 2x2 maxpool -> phiT[16][64][1024], Gp[16][1024][256] ----
__global__ void pool_kernel(const u16* __restrict__ Y, u16* __restrict__ phiT,
                            u16* __restrict__ Gp) {
  int idx = blockIdx.x * 256 + threadIdx.x;
  const int PHI_N = 16 * 1024 * 64;
  if (idx < PHI_N) {
    int c = idx & 63;
    int m = (idx >> 6) & 1023;
    int b = idx >> 16;
    int h2 = m >> 5, w2 = m & 31;
    size_t r0 = (size_t)b * 4096 + (size_t)(h2 * 2) * 64 + w2 * 2;
    const u16* p = Y + r0 * 384 + 64 + c;
    float mv = fmaxf(fmaxf(b2f(p[0]), b2f(p[384])),
                     fmaxf(b2f(p[64 * 384]), b2f(p[65 * 384])));
    phiT[((size_t)b * 64 + c) * 1024 + m] = f2b(mv);   // transposed: [kd][key]
  } else {
    int j = idx - PHI_N;
    if (j < 16 * 1024 * 256) {
      int c = j & 255;
      int m = (j >> 8) & 1023;
      int b = j >> 18;
      int h2 = m >> 5, w2 = m & 31;
      size_t r0 = (size_t)b * 4096 + (size_t)(h2 * 2) * 64 + w2 * 2;
      const u16* p = Y + r0 * 384 + 128 + c;
      float mv = fmaxf(fmaxf(b2f(p[0]), b2f(p[384])),
                       fmaxf(b2f(p[64 * 384]), b2f(p[65 * 384])));
      Gp[((size_t)b * 1024 + m) * 256 + c] = f2b(mv);  // row-major: [key][ch]
    }
  }
}

// --------- Kernel 4: fused attention (scores -> softmax in LDS -> P@g) -------
__global__ void attn_kernel(const u16* __restrict__ Y, const u16* __restrict__ phiT,
                            const u16* __restrict__ Gp, u16* __restrict__ AG) {
  extern __shared__ float lds[];
  int lane = threadIdx.x & 31;
  int wid  = threadIdx.x >> 5;
  float* S    = lds + (size_t)wid * (16 * 1024 + 16);
  float* sums = S + 16 * 1024;
  int wave = blockIdx.x * 2 + wid;
  int b  = wave >> 8;
  int rb = wave & 255;
  if (b >= 16) return;

  const u16* Q = Y + (size_t)(b * 4096 + rb * 16) * 384;  // theta = cols 0..63 of Y
  bf16x16 qa0 = load_a_bf16(Q,      384, lane);
  bf16x16 qa1 = load_a_bf16(Q + 32, 384, lane);
  const u16* P = phiT + (size_t)b * 64 * 1024;

  int n = lane & 15, hh = lane >> 4;
  f32x8 z = {0.f,0.f,0.f,0.f,0.f,0.f,0.f,0.f};
  // Phase 1: scores S[16][1024] = theta @ phi^T
  for (int nt = 0; nt < 64; ++nt) {
    f32x8 acc = z;
    bf16x16 b0 = load_b_bf16(P + nt * 16, 1024, lane);
    acc = wmma_bf16(qa0, b0, acc);
    bf16x16 b1 = load_b_bf16(P + (size_t)32 * 1024 + nt * 16, 1024, lane);
    acc = wmma_bf16(qa1, b1, acc);
#pragma unroll
    for (int v = 0; v < 8; ++v) S[(v + 8 * hh) * 1024 + nt * 16 + n] = acc[v];
  }
  __syncthreads();

  // Phase 2: row softmax (unnormalized exp in place); lanes L,L^16 share row L&15
  int r = lane & 15;
  f32x4* rowp = (f32x4*)(S + r * 1024 + hh * 512);
  float mx = -3.0e38f;
  for (int i = 0; i < 128; ++i) {
    f32x4 t = rowp[i];
    mx = fmaxf(mx, fmaxf(fmaxf(t.x, t.y), fmaxf(t.z, t.w)));
  }
  mx = fmaxf(mx, __shfl_xor(mx, 16, 32));
  float sm = 0.f;
  for (int i = 0; i < 128; ++i) {
    f32x4 t = rowp[i];
    t.x = __expf(t.x - mx); t.y = __expf(t.y - mx);
    t.z = __expf(t.z - mx); t.w = __expf(t.w - mx);
    sm += t.x + t.y + t.z + t.w;
    rowp[i] = t;
  }
  sm += __shfl_xor(sm, 16, 32);
  if (hh == 0) sums[r] = sm;
  __syncthreads();

  float invs[8];
#pragma unroll
  for (int v = 0; v < 8; ++v) invs[v] = 1.0f / sums[v + 8 * hh];

  // Phase 3: O = (expS @ g) / sum  -- two passes of 8 channel tiles (acc[8])
  const u16* G = Gp + (size_t)b * 1024 * 256;
  u16* O = AG + (size_t)(b * 4096 + rb * 16) * 256;
  for (int co = 0; co < 2; ++co) {
    f32x8 acc[8];
#pragma unroll
    for (int c = 0; c < 8; ++c) acc[c] = z;
    for (int kt = 0; kt < 32; ++kt) {               // K = 1024 keys
      bf16x16 a = load_a_f32(S + kt * 32, 1024, lane);
#pragma unroll
      for (int c = 0; c < 8; ++c) {
        bf16x16 bb = load_b_bf16(G + (size_t)(kt * 32) * 256 + (co * 8 + c) * 16,
                                 256, lane);
        acc[c] = wmma_bf16(a, bb, acc[c]);
      }
    }
#pragma unroll
    for (int c = 0; c < 8; ++c) {
#pragma unroll
      for (int v = 0; v < 8; ++v) {
        O[(size_t)(v + 8 * hh) * 256 + (co * 8 + c) * 16 + n] =
            f2b(acc[c][v] * invs[v]);
      }
    }
  }
}

// --------- Kernel 5: out = x + sigma * (attn_g @ W_attn) ---------------------
// One wave = 16-row strip x 8 N-tiles (quarter of 512 cols); acc[8] = 64 VGPRs.
__global__ void final_gemm(const u16* __restrict__ AG, const u16* __restrict__ W,
                           const float* __restrict__ x, const float* __restrict__ sigma,
                           float* __restrict__ out) {
  int gw   = (int)((blockIdx.x * blockDim.x + threadIdx.x) >> 5);
  int lane = threadIdx.x & 31;
  int mt   = gw >> 2;          // 0..4095
  int q    = gw & 3;           // N quarter
  const u16* Ab = AG + (size_t)mt * 16 * 256;

  f32x8 z = {0.f,0.f,0.f,0.f,0.f,0.f,0.f,0.f};
  f32x8 acc[8];
#pragma unroll
  for (int c = 0; c < 8; ++c) acc[c] = z;

  for (int kt = 0; kt < 8; ++kt) {                  // K = 256
    bf16x16 a = load_a_bf16(Ab + kt * 32, 256, lane);
#pragma unroll
    for (int c = 0; c < 8; ++c) {
      bf16x16 b = load_b_bf16(W + (size_t)(kt * 32) * 512 + (q * 8 + c) * 16,
                              512, lane);
      acc[c] = wmma_bf16(a, b, acc[c]);
    }
  }
  float sg = sigma[0];
  int n = lane & 15, hh = lane >> 4;
#pragma unroll
  for (int c = 0; c < 8; ++c) {
#pragma unroll
    for (int v = 0; v < 8; ++v) {
      size_t idx = ((size_t)mt * 16 + v + 8 * hh) * 512 + (q * 8 + c) * 16 + n;
      out[idx] = x[idx] + sg * acc[c][v];
    }
  }
}

extern "C" void kernel_launch(void* const* d_in, const int* in_sizes, int n_in,
                              void* d_out, int out_size, void* d_ws, size_t ws_size,
                              hipStream_t stream) {
  (void)in_sizes; (void)n_in; (void)out_size; (void)ws_size;
  const float* x       = (const float*)d_in[0];
  const float* k_theta = (const float*)d_in[1];
  const float* u_theta = (const float*)d_in[2];
  const float* k_phi   = (const float*)d_in[3];
  const float* u_phi   = (const float*)d_in[4];
  const float* k_g     = (const float*)d_in[5];
  const float* u_g     = (const float*)d_in[6];
  const float* k_attn  = (const float*)d_in[7];
  const float* u_attn  = (const float*)d_in[8];
  const float* sigma   = (const float*)d_in[9];
  float* out = (float*)d_out;

  char* ws = (char*)d_ws;
  size_t off = 0;
  auto alloc = [&](size_t bytes) -> void* {
    void* p = ws + off;
    off = (off + bytes + 255) & ~(size_t)255;
    return p;
  };
  u16* Wcat  = (u16*)alloc((size_t)512 * 384 * 2);   // [theta|phi|g] concatenated
  u16* wn_a  = (u16*)alloc((size_t)256 * 512 * 2);
  u16* Y     = (u16*)alloc((size_t)65536 * 384 * 2);
  u16* phiT  = (u16*)alloc((size_t)16 * 64 * 1024 * 2);
  u16* Gp    = (u16*)alloc((size_t)16 * 1024 * 256 * 2);
  u16* AG    = (u16*)alloc((size_t)65536 * 256 * 2);

  spectral_pack<<<1, 256, 0, stream>>>(k_theta, u_theta, Wcat, 512, 64,  384, 0);
  spectral_pack<<<1, 256, 0, stream>>>(k_phi,   u_phi,   Wcat, 512, 64,  384, 64);
  spectral_pack<<<1, 256, 0, stream>>>(k_g,     u_g,     Wcat, 512, 256, 384, 128);
  spectral_pack<<<1, 256, 0, stream>>>(k_attn,  u_attn,  wn_a, 256, 512, 512, 0);

  // 8192 waves (4096 strips x 2 N-halves), 8 waves / block
  conv_gemm<<<1024, 256, 0, stream>>>(x, Wcat, Y);

  pool_kernel<<<20480, 256, 0, stream>>>(Y, phiT, Gp);

  // 4096 waves, 2 per block; dynamic LDS: 2 * (16*1024 + 16) floats
  attn_kernel<<<2048, 64, 2 * (16 * 1024 + 16) * sizeof(float), stream>>>(Y, phiT, Gp, AG);

  // 16384 waves (4096 strips x 4 N-quarters), 8 waves / block
  final_gemm<<<2048, 256, 0, stream>>>(AG, wn_a, x, sigma, out);
}
